// PhyGaussNO_32719060861307
// MI455X (gfx1250) — matrix operand order repaired and checked by reading, not compiled, not run on hardware
//
#include <hip/hip_runtime.h>
#include <hip/hip_bf16.h>

// Problem constants (from reference)
#define Bn 4
#define Nn 16384
#define En 262144
#define Dd 2
#define Cc 128
#define Kk 256
#define NLn 3
#define FCn 128

typedef _Float16 v16h __attribute__((ext_vector_type(16)));
typedef _Float16 v8h  __attribute__((ext_vector_type(8)));
typedef float    v8f  __attribute__((ext_vector_type(8)));

__device__ __forceinline__ float gelu_tanh(float x) {
    float x3 = x * x * x;
    float t = tanhf(0.7978845608028654f * (x + 0.044715f * x3));
    return 0.5f * x * (1.0f + t);
}

__device__ __forceinline__ v8f wmma_f16(v16h a, v16h b, v8f c) {
    // 8 args: (neg_a, A, neg_b, B, c_mod, C, reuse_a, reuse_b)
    return __builtin_amdgcn_wmma_f32_16x16x32_f16(false, a, false, b, (short)0, c, false, false);
}

// A fragment (16x32 f16): lane m = lane&15; lanes 0-15 hold K 0..15,
// lanes 16-31 hold K 16..31 (contiguous, matching B).
__device__ __forceinline__ v16h load_fragA(const _Float16* __restrict__ rowA, int kb8) {
    v8h lo = *(const v8h*)(rowA + 2 * kb8);
    v8h hi = *(const v8h*)(rowA + 2 * kb8 + 8);
    return __builtin_shufflevector(lo, hi, 0, 1, 2, 3, 4, 5, 6, 7, 8, 9, 10, 11, 12, 13, 14, 15);
}

// ---------------------------------------------------------------------------
// Generic NT GEMM: C[m,n] = alpha * sum_k A[m,k]*B[n,k] + brow[m] + bcol[n]
// One wave per 16x16 output tile, 4 waves per block sharing tile_m.
// The shared A tile is staged into LDS with the CDNA5 async-copy engine
// (GLOBAL_LOAD_ASYNC_TO_LDS_B128, ASYNCcnt double-buffering); the per-wave
// B fragments stay on direct pipelined global_load_b128.
// ---------------------------------------------------------------------------
__global__ void gemm_nt_wmma(const _Float16* __restrict__ A,
                             const _Float16* __restrict__ Bm,
                             float* __restrict__ Cout,
                             int lda, int ldb, int ldc, int Kc,
                             long long sA, long long sB, long long sC,
                             float alpha,
                             const float* __restrict__ brow,
                             const float* __restrict__ bcol) {
    int bz = blockIdx.z;
    A    += (long long)bz * sA;
    Bm   += (long long)bz * sB;
    Cout += (long long)bz * sC;

    int lane = threadIdx.x & 31;
    int wave = threadIdx.x >> 5;
    int tile_n = (blockIdx.x * 4 + wave) * 16;
    int tile_m = blockIdx.y * 16;
    int am   = lane & 15;
    int kb8  = ((lane >> 4) & 1) * 8;
    int kb16 = ((lane >> 4) & 1) * 16;

    __shared__ _Float16 As[2][16 * 32];   // double-buffered A tile (2 x 1KB)

    const _Float16* Abase = A + (long long)tile_m * lda;
    const _Float16* rowB  = Bm + (long long)(tile_n + am) * ldb;

    int t = threadIdx.x;
    // Waves 0-1 (threads 0..63) cooperatively async-copy one 16x32 f16 tile:
    // 64 lanes x 16B = 1KB, lane-addressed scatter into LDS.
    auto stageA = [&](int buf, int k0) {
        if (t < 64) {
            int m  = t >> 2;
            int ch = t & 3;
            const _Float16* g = Abase + (long long)m * lda + k0 + ch * 8;
            unsigned lds = (unsigned)(unsigned long long)(uintptr_t)&As[buf][m * 32 + ch * 8];
            asm volatile("global_load_async_to_lds_b128 %0, %1, off"
                         :: "v"(lds), "v"(g) : "memory");
        }
    };

    stageA(0, 0);

    v8f acc = {0.f, 0.f, 0.f, 0.f, 0.f, 0.f, 0.f, 0.f};
    for (int k0 = 0; k0 < Kc; k0 += 32) {
        int buf = (k0 >> 5) & 1;
        bool more = (k0 + 32) < Kc;
        if (more) {
            stageA(buf ^ 1, k0 + 32);
            // keep the just-issued copy in flight; in-order completion means
            // asynccnt<=1 implies the current buffer has landed
            asm volatile("s_wait_asynccnt 1" ::: "memory");
        } else {
            asm volatile("s_wait_asynccnt 0" ::: "memory");
        }
        __syncthreads();

        __builtin_prefetch(rowB + k0 + 512, 0, 1);   // speculative, OOB-safe
        v8h lo = *(const v8h*)&As[buf][am * 32 + kb16];
        v8h hi = *(const v8h*)&As[buf][am * 32 + kb16 + 8];
        v16h a = __builtin_shufflevector(lo, hi, 0, 1, 2, 3, 4, 5, 6, 7,
                                         8, 9, 10, 11, 12, 13, 14, 15);
        v16h b = *(const v16h*)(rowB + k0 + kb16);
        acc = wmma_f16(a, b, acc);

        __syncthreads();   // reads done before this buffer is refilled
    }

    int cn = tile_n + (lane & 15);
    int mh = (lane >> 4) & 1;
    float cb = bcol ? bcol[cn] : 0.f;
    for (int r = 0; r < 8; ++r) {
        int cm = tile_m + r + mh * 8;
        float v = alpha * acc[r] + cb + (brow ? brow[cm] : 0.f);
        Cout[(long long)cm * ldc + cn] = v;
    }
}

// ---------------------------------------------------------------------------
// Fused inverse spectral transform (Kc=256 only, A tiles are L2-hot):
// x1[b,o,n] = 2*( fc[b,o,:]·bc[b,n,:] + fs[b,o,:]·bs[b,n,:] ) + f0[b,o]*mask[b,n]
// ---------------------------------------------------------------------------
__global__ void gemm_inv_fused(const _Float16* __restrict__ fc16,
                               const _Float16* __restrict__ fs16,
                               const _Float16* __restrict__ bc16,
                               const _Float16* __restrict__ bs16,
                               const float* __restrict__ f0,
                               const float* __restrict__ mask,
                               float* __restrict__ x1) {
    int b = blockIdx.z;
    const _Float16* Ac = fc16 + (long long)b * Cc * Kk;
    const _Float16* As = fs16 + (long long)b * Cc * Kk;
    const _Float16* Bc = bc16 + (long long)b * Nn * Kk;
    const _Float16* Bs = bs16 + (long long)b * Nn * Kk;

    int lane = threadIdx.x & 31;
    int wave = threadIdx.x >> 5;
    int tile_n = (blockIdx.x * 4 + wave) * 16;
    int tile_m = blockIdx.y * 16;
    int am   = lane & 15;
    int kb8  = ((lane >> 4) & 1) * 8;
    int kb16 = ((lane >> 4) & 1) * 16;

    const _Float16* rAc = Ac + (long long)(tile_m + am) * Kk;
    const _Float16* rAs = As + (long long)(tile_m + am) * Kk;
    const _Float16* rBc = Bc + (long long)(tile_n + am) * Kk;
    const _Float16* rBs = Bs + (long long)(tile_n + am) * Kk;

    v8f acc = {0.f, 0.f, 0.f, 0.f, 0.f, 0.f, 0.f, 0.f};
    for (int k0 = 0; k0 < Kk; k0 += 32) {
        v16h a = load_fragA(rAc + k0, kb8);
        v16h bb = *(const v16h*)(rBc + k0 + kb16);
        acc = wmma_f16(a, bb, acc);
        a = load_fragA(rAs + k0, kb8);
        bb = *(const v16h*)(rBs + k0 + kb16);
        acc = wmma_f16(a, bb, acc);
    }

    int cn = tile_n + (lane & 15);
    int mh = (lane >> 4) & 1;
    float mk = mask[(long long)b * Nn + cn];
    for (int r = 0; r < 8; ++r) {
        int cm = tile_m + r + mh * 8;
        x1[((long long)(b * Cc + cm)) * Nn + cn] = 2.f * acc[r] + f0[b * Cc + cm] * mk;
    }
}

// ---------------------------------------------------------------------------
// Fourier bases: bc/bs (B,N,K) for the inverse pass (k-fastest), weighted
// bases transposed (B,K,N) for the forward pass (n-fastest).
// ---------------------------------------------------------------------------
__global__ void bases_kernel(const float* __restrict__ nodes, const float* __restrict__ modes,
                             const float* __restrict__ mask, const float* __restrict__ nw,
                             _Float16* __restrict__ bc16, _Float16* __restrict__ bs16,
                             _Float16* __restrict__ wbcT, _Float16* __restrict__ wbsT) {
    long long bn = blockIdx.x;
    int k = threadIdx.x;
    int b = (int)(bn / Nn), n = (int)(bn % Nn);
    float nx = nodes[bn * 2], ny = nodes[bn * 2 + 1];
    float t = nx * modes[k * 2] + ny * modes[k * 2 + 1];
    float cv = cosf(t), sv = sinf(t);
    float m = mask[bn], w = nw[bn];
    bc16[bn * Kk + k] = (_Float16)(cv * m);
    bs16[bn * Kk + k] = (_Float16)(sv * m);
    wbcT[(((long long)b * Kk) + k) * Nn + n] = (_Float16)(cv * m * w);
    wbsT[(((long long)b * Kk) + k) * Nn + n] = (_Float16)(sv * m * w);
}

// h = x @ fc0_w + fc0_b ; write all three layouts of h
__global__ void fc0_kernel(const float* __restrict__ x, const float* __restrict__ w,
                           const float* __restrict__ bias,
                           _Float16* __restrict__ h16, _Float16* __restrict__ hT16,
                           float* __restrict__ hT32) {
    long long bn = blockIdx.x;
    int c = threadIdx.x;
    int b = (int)(bn / Nn), n = (int)(bn % Nn);
    float v = bias[c];
    for (int i = 0; i < 3; ++i) v += x[bn * 3 + i] * w[i * Cc + c];
    h16[((long long)(b * Cc + c)) * Nn + n] = (_Float16)v;
    hT16[bn * Cc + c] = (_Float16)v;
    hT32[bn * Cc + c] = v;
}

// x0[b,c] = sum_n h[b,c,n] * nw * mask  (LDS tree reduction)
__global__ void x0_reduce(const float* __restrict__ hT32, const float* __restrict__ nw,
                          const float* __restrict__ mask, float* __restrict__ x0) {
    int b = blockIdx.x >> 7;
    int c = blockIdx.x & 127;
    __shared__ float red[256];
    float s = 0.f;
    for (int n = threadIdx.x; n < Nn; n += 256) {
        long long bn = (long long)b * Nn + n;
        s += hT32[bn * Cc + c] * nw[bn] * mask[bn];
    }
    red[threadIdx.x] = s;
    __syncthreads();
    for (int st = 128; st > 0; st >>= 1) {
        if ((int)threadIdx.x < st) red[threadIdx.x] += red[threadIdx.x + st];
        __syncthreads();
    }
    if (threadIdx.x == 0) x0[b * Cc + c] = red[0];
}

// fc/fs mode mixing over channels (small: 134 MFLOP total)
__global__ void fcfs_kernel(const float* __restrict__ xc, const float* __restrict__ xs,
                            const float* __restrict__ wc, const float* __restrict__ ws,
                            _Float16* __restrict__ fc16, _Float16* __restrict__ fs16) {
    int k = threadIdx.x;
    int o = blockIdx.x & 127;
    int b = blockIdx.x >> 7;
    float fcv = 0.f, fsv = 0.f;
    for (int c = 0; c < Cc; ++c) {
        float xcv = xc[((long long)(b * Cc + c)) * Kk + k];
        float xsv = xs[((long long)(b * Cc + c)) * Kk + k];
        float wcv = wc[((long long)(c * Cc + o)) * Kk + k];
        float wsv = ws[((long long)(c * Cc + o)) * Kk + k];
        fcv += xcv * wcv - xsv * wsv;
        fsv += xsv * wcv + xcv * wsv;
    }
    fc16[((long long)(b * Cc + o)) * Kk + k] = (_Float16)fcv;
    fs16[((long long)(b * Cc + o)) * Kk + k] = (_Float16)fsv;
}

__global__ void f0_kernel(const float* __restrict__ x0, const float* __restrict__ w0,
                          float* __restrict__ f0) {
    int i = blockIdx.x * blockDim.x + threadIdx.x;
    if (i >= Bn * Cc) return;
    int b = i >> 7, o = i & 127;
    float s = 0.f;
    for (int c = 0; c < Cc; ++c) s += x0[b * Cc + c] * w0[c * Cc + o];
    f0[i] = s;
}

__global__ void f32_to_f16(const float* __restrict__ src, _Float16* __restrict__ dst, long long n) {
    long long i = (long long)blockIdx.x * blockDim.x + threadIdx.x;
    if (i < n) dst[i] = (_Float16)src[i];
}

// src (R rows, Ccols cols) -> dst (Ccols, R) in f16
__global__ void transpose_to_f16(const float* __restrict__ src, _Float16* __restrict__ dst,
                                 int R, int Ccols) {
    int i = blockIdx.x * blockDim.x + threadIdx.x;
    if (i < R * Ccols) {
        int r = i / Ccols, c = i % Ccols;
        dst[c * R + r] = (_Float16)src[i];
    }
}

__global__ void zero_f32(float* __restrict__ p, long long n) {
    long long i = (long long)blockIdx.x * blockDim.x + threadIdx.x;
    long long stride = (long long)gridDim.x * blockDim.x;
    for (; i < n; i += stride) p[i] = 0.f;
}

// Scatter edge-gradient messages:  g[b,tgt,d*C+c] += egw[b,e,d]*(h[src,c]-h[tgt,c])
__global__ void edge_grad(const float* __restrict__ hT32, const int* __restrict__ edges,
                          const float* __restrict__ egw, float* __restrict__ g32) {
    long long be = blockIdx.x;
    int b = (int)(be / En);
    int src = edges[be * 2 + 0], tgt = edges[be * 2 + 1];
    int dc = threadIdx.x;
    int d = dc >> 7, c = dc & 127;
    float df = hT32[((long long)b * Nn + src) * Cc + c] - hT32[((long long)b * Nn + tgt) * Cc + c];
    float w = egw[be * 2 + d];
    atomicAdd(&g32[((long long)b * Nn + tgt) * (Dd * Cc) + dc], w * df);
}

// Self-edge init: out = f*nw, denom = nw  (ef=0 -> exp(0)=1)
__global__ void scat_init(const float* __restrict__ fgr, const float* __restrict__ nw,
                          float* __restrict__ scat, float* __restrict__ denom) {
    long long bn = blockIdx.x;
    int c = threadIdx.x;
    float w = nw[bn];
    scat[bn * Cc + c] = fgr[bn * Cc + c] * w;
    if (c == 0) denom[bn] = w;
}

// Gaussian-kernel edge aggregation
__global__ void edge_scatter(const float* __restrict__ fgr, const float* __restrict__ nodes,
                             const float* __restrict__ nw, const int* __restrict__ edges,
                             const float* __restrict__ cholL, const float* __restrict__ cholb,
                             float* __restrict__ scat, float* __restrict__ denom) {
    long long be = blockIdx.x;
    int b = (int)(be / En);
    int src = edges[be * 2 + 0], tgt = edges[be * 2 + 1];
    long long is = (long long)b * Nn + src, it = (long long)b * Nn + tgt;
    float ef0 = nodes[is * 2] - nodes[it * 2];
    float ef1 = nodes[is * 2 + 1] - nodes[it * 2 + 1];
    float l00 = cholL[0], l01 = cholL[1], l10 = cholL[2], l11 = cholL[3];
    float w00 = l00 * l00 + l01 * l01;
    float w01 = l00 * l10 + l01 * l11;
    float w11 = l10 * l10 + l11 * l11;
    float v0 = w00 * ef0 + w01 * ef1 + cholb[0];
    float v1 = w01 * ef0 + w11 * ef1 + cholb[1];
    float q = ef0 * v0 + ef1 * v1;
    float w = nw[is] * __expf(-q);
    int c = threadIdx.x;
    atomicAdd(&scat[it * Cc + c], fgr[is * Cc + c] * w);
    if (c == 0) atomicAdd(&denom[it], w);
}

// h_{i+1} = gelu(x1+x2) + gelu(scat/denom)  (or x1+x2 at last layer)
__global__ void update_h(const float* __restrict__ x1, const float* __restrict__ x2,
                         const float* __restrict__ scat, const float* __restrict__ denom,
                         _Float16* __restrict__ h16, _Float16* __restrict__ hT16,
                         float* __restrict__ hT32, int use_graph) {
    long long bn = blockIdx.x;
    int b = (int)(bn / Nn), n = (int)(bn % Nn);
    int o = threadIdx.x;
    float v = x1[((long long)(b * Cc + o)) * Nn + n] + x2[((long long)(b * Cc + o)) * Nn + n];
    if (use_graph) {
        float g3 = scat[bn * Cc + o] / denom[bn];
        v = gelu_tanh(v) + gelu_tanh(g3);
    }
    h16[((long long)(b * Cc + o)) * Nn + n] = (_Float16)v;
    hT16[bn * Cc + o] = (_Float16)v;
    hT32[bn * Cc + o] = v;
}

// out[b,n] = gelu(hfc[b,n,:]) . fc2_w + fc2_b
__global__ void final_out(const float* __restrict__ hfc, const float* __restrict__ fc2w,
                          const float* __restrict__ fc2b, float* __restrict__ out) {
    long long i = (long long)blockIdx.x * blockDim.x + threadIdx.x;
    if (i >= (long long)Bn * Nn) return;
    float s = fc2b[0];
    const float* row = hfc + i * FCn;
    for (int f = 0; f < FCn; ++f) s += gelu_tanh(row[f]) * fc2w[f];
    out[i] = s;
}

// ---------------------------------------------------------------------------
extern "C" void kernel_launch(void* const* d_in, const int* in_sizes, int n_in,
                              void* d_out, int out_size, void* d_ws, size_t ws_size,
                              hipStream_t stream) {
    // Inputs flattened in setup_inputs() order (params dict leaves in order):
    const float* x        = (const float*)d_in[0];
    const float* mask     = (const float*)d_in[1];
    const float* nodes    = (const float*)d_in[2];
    const float* nw       = (const float*)d_in[3];
    const float* egw      = (const float*)d_in[4];
    const float* modes    = (const float*)d_in[5];
    const float* fc0_w    = (const float*)d_in[6];
    const float* fc0_b    = (const float*)d_in[7];
    const float* sp_wc    = (const float*)d_in[8];
    const float* sp_ws    = (const float*)d_in[9];
    const float* sp_w0    = (const float*)d_in[10];
    const float* w_w      = (const float*)d_in[11];
    const float* w_b      = (const float*)d_in[12];
    const float* mlp_w    = (const float*)d_in[13];
    const float* mlp_b    = (const float*)d_in[14];
    const float* chol_L   = (const float*)d_in[15];
    const float* chol_b   = (const float*)d_in[16];
    const float* fc1_w    = (const float*)d_in[17];
    const float* fc1_b    = (const float*)d_in[18];
    const float* fc2_w    = (const float*)d_in[19];
    const float* fc2_b    = (const float*)d_in[20];
    const int*   edges    = (const int*)d_in[21];
    float* out = (float*)d_out;

    // Workspace bump allocator (256B aligned)
    char* ws = (char*)d_ws;
    auto alloc = [&](size_t bytes) -> void* {
        void* p = ws;
        ws += (bytes + 255) & ~(size_t)255;
        return p;
    };
    _Float16* bc16   = (_Float16*)alloc((size_t)Bn * Nn * Kk * 2);
    _Float16* bs16   = (_Float16*)alloc((size_t)Bn * Nn * Kk * 2);
    _Float16* wbcT   = (_Float16*)alloc((size_t)Bn * Kk * Nn * 2);
    _Float16* wbsT   = (_Float16*)alloc((size_t)Bn * Kk * Nn * 2);
    _Float16* h16    = (_Float16*)alloc((size_t)Bn * Cc * Nn * 2);
    _Float16* hT16   = (_Float16*)alloc((size_t)Bn * Nn * Cc * 2);
    float*    hT32   = (float*)alloc((size_t)Bn * Nn * Cc * 4);
    float*    xc     = (float*)alloc((size_t)Bn * Cc * Kk * 4);
    float*    xsv    = (float*)alloc((size_t)Bn * Cc * Kk * 4);
    float*    x0     = (float*)alloc((size_t)Bn * Cc * 4);
    _Float16* fc16   = (_Float16*)alloc((size_t)Bn * Cc * Kk * 2);
    _Float16* fs16   = (_Float16*)alloc((size_t)Bn * Cc * Kk * 2);
    float*    f0     = (float*)alloc((size_t)Bn * Cc * 4);
    float*    x1     = (float*)alloc((size_t)Bn * Cc * Nn * 4);
    float*    x2     = (float*)alloc((size_t)Bn * Cc * Nn * 4);
    _Float16* wconv16 = (_Float16*)alloc((size_t)Cc * Cc * 2);
    _Float16* mlpw16  = (_Float16*)alloc((size_t)Cc * Dd * Cc * 2);
    float*    g32    = (float*)alloc((size_t)Bn * Nn * Dd * Cc * 4);
    _Float16* g16    = (_Float16*)alloc((size_t)Bn * Nn * Dd * Cc * 2);
    float*    fgr    = (float*)alloc((size_t)Bn * Nn * Cc * 4);
    float*    scat   = (float*)alloc((size_t)Bn * Nn * Cc * 4);
    float*    denom  = (float*)alloc((size_t)Bn * Nn * 4);
    _Float16* fc1wT16 = (_Float16*)alloc((size_t)FCn * Cc * 2);
    float*    hfc    = (float*)alloc((size_t)Bn * Nn * FCn * 4);

    // 1. Fourier bases + weighted transposed bases
    bases_kernel<<<Bn * Nn, Kk, 0, stream>>>(nodes, modes, mask, nw, bc16, bs16, wbcT, wbsT);

    // 2. fc0 lifting
    fc0_kernel<<<Bn * Nn, Cc, 0, stream>>>(x, fc0_w, fc0_b, h16, hT16, hT32);

    for (int i = 0; i < NLn; ++i) {
        const float* sp_wc_i = sp_wc + (size_t)i * Cc * Cc * Kk;
        const float* sp_ws_i = sp_ws + (size_t)i * Cc * Cc * Kk;
        const float* sp_w0_i = sp_w0 + (size_t)i * Cc * Cc;
        const float* w_w_i   = w_w + (size_t)i * Cc * Cc;
        const float* w_b_i   = w_b + (size_t)i * Cc;
        const float* mlp_w_i = mlp_w + (size_t)i * Cc * Dd * Cc;
        const float* mlp_b_i = mlp_b + (size_t)i * Cc;
        const float* cholL_i = chol_L + (size_t)i * 4;
        const float* cholb_i = chol_b + (size_t)i * 2;

        // Forward spectral transform (contraction over N=16384)
        gemm_nt_wmma<<<dim3(Kk / 64, Cc / 16, Bn), 128, 0, stream>>>(
            h16, wbcT, xc, Nn, Nn, Kk, Nn,
            (long long)Cc * Nn, (long long)Kk * Nn, (long long)Cc * Kk,
            1.0f, nullptr, nullptr);
        gemm_nt_wmma<<<dim3(Kk / 64, Cc / 16, Bn), 128, 0, stream>>>(
            h16, wbsT, xsv, Nn, Nn, Kk, Nn,
            (long long)Cc * Nn, (long long)Kk * Nn, (long long)Cc * Kk,
            -1.0f, nullptr, nullptr);   // xs = -einsum(...)

        // DC component
        x0_reduce<<<Bn * Cc, 256, 0, stream>>>(hT32, nw, mask, x0);

        // Mode mixing
        fcfs_kernel<<<Bn * Cc, Kk, 0, stream>>>(xc, xsv, sp_wc_i, sp_ws_i, fc16, fs16);
        f0_kernel<<<2, 256, 0, stream>>>(x0, sp_w0_i, f0);

        // Inverse spectral transform fused with f0*mask -> x1
        gemm_inv_fused<<<dim3(Nn / 64, Cc / 16, Bn), 128, 0, stream>>>(
            fc16, fs16, bc16, bs16, f0, mask, x1);

        // Pointwise channel mix x2 = W h + b
        f32_to_f16<<<(Cc * Cc + 255) / 256, 256, 0, stream>>>(w_w_i, wconv16, (long long)Cc * Cc);
        gemm_nt_wmma<<<dim3(Nn / 64, Cc / 16, Bn), 128, 0, stream>>>(
            wconv16, hT16, x2, Cc, Cc, Nn, Cc,
            0LL, (long long)Nn * Cc, (long long)Cc * Nn,
            1.0f, w_b_i, nullptr);

        if (i != NLn - 1) {
            // --- phy_graph_conv ---
            zero_f32<<<4096, 256, 0, stream>>>(g32, (long long)Bn * Nn * Dd * Cc);
            edge_grad<<<Bn * En, Dd * Cc, 0, stream>>>(hT32, edges, egw, g32);
            f32_to_f16<<<(int)(((long long)Bn * Nn * Dd * Cc + 255) / 256), 256, 0, stream>>>(
                g32, g16, (long long)Bn * Nn * Dd * Cc);
            f32_to_f16<<<(Cc * Dd * Cc + 255) / 256, 256, 0, stream>>>(
                mlp_w_i, mlpw16, (long long)Cc * Dd * Cc);
            // f[b,n,o] = g[b,n,:] . mlp_w[o,:] + mlp_b[o]
            gemm_nt_wmma<<<dim3(Cc / 64, Nn / 16, Bn), 128, 0, stream>>>(
                g16, mlpw16, fgr, Dd * Cc, Dd * Cc, Cc, Dd * Cc,
                (long long)Nn * Dd * Cc, 0LL, (long long)Nn * Cc,
                1.0f, nullptr, mlp_b_i);
            // Self edges, then Gaussian-kernel edge aggregation
            scat_init<<<Bn * Nn, Cc, 0, stream>>>(fgr, nw, scat, denom);
            edge_scatter<<<Bn * En, Cc, 0, stream>>>(fgr, nodes, nw, edges, cholL_i, cholb_i,
                                                     scat, denom);
            update_h<<<Bn * Nn, Cc, 0, stream>>>(x1, x2, scat, denom, h16, hT16, hT32, 1);
        } else {
            update_h<<<Bn * Nn, Cc, 0, stream>>>(x1, x2, scat, denom, h16, hT16, hT32, 0);
        }
    }

    // Projection head: gelu(h @ fc1 + b1) @ fc2 + b2
    transpose_to_f16<<<(Cc * FCn + 255) / 256, 256, 0, stream>>>(fc1_w, fc1wT16, Cc, FCn);
    gemm_nt_wmma<<<dim3(FCn / 64, Nn / 16, Bn), 128, 0, stream>>>(
        hT16, fc1wT16, hfc, Cc, Cc, FCn, Cc,
        (long long)Nn * Cc, 0LL, (long long)Nn * FCn,
        1.0f, nullptr, fc1_b);
    final_out<<<(Bn * Nn + 255) / 256, 256, 0, stream>>>(hfc, fc2_w, fc2_b, out);
}
